// KmerDPVAE_12962211299863
// MI455X (gfx1250) — compile-verified, matrix operand order checked by reading, not run on hardware
//
#include <hip/hip_runtime.h>
#include <hip/hip_bf16.h>

typedef __attribute__((ext_vector_type(16))) __bf16        v16bf;
typedef __attribute__((ext_vector_type(8)))  float         v8f;
typedef __attribute__((ext_vector_type(4)))  unsigned int  v4u;

#define NN   4096
#define KK   500
#define KPAD 512
#define DZ   64
#define ALPHA_C 1.0f
#define SMIN_C  1e-6f

// ---------------------------------------------------------------- digamma
__device__ __forceinline__ float digammaf_dev(float x) {
    float r = 0.f;
    while (x < 6.f) { r -= 1.f / x; x += 1.f; }
    float inv  = 1.f / x;
    float inv2 = inv * inv;
    float s = __logf(x) - 0.5f * inv
            - inv2 * (1.f/12.f - inv2 * (1.f/120.f - inv2 * (1.f/252.f)));
    return r + s;
}

// ---------------------------------------------------------------- E[log pi]
__global__ void elogpi_kernel(const float* __restrict__ a,
                              const float* __restrict__ b,
                              float* __restrict__ elp) {
    if (threadIdx.x != 0 || blockIdx.x != 0) return;
    float cs = 0.f;
    for (int k = 0; k < KK; ++k) {
        float ak = a[k], bk = b[k];
        float pab = digammaf_dev(ak + bk);
        float pa  = digammaf_dev(ak);
        float pb  = digammaf_dev(bk);
        elp[k] = (pa - pab) + cs;
        cs    += (pb - pab);
    }
}

// ---------------------------------------------------------------- E-step (fused log-gauss + softmax per row)
__global__ void __launch_bounds__(256)
estep_kernel(const float* __restrict__ mu, const float* __restrict__ s2,
             const float* __restrict__ S,  const float* __restrict__ m,
             const float* __restrict__ elp, float* __restrict__ r) {
    __shared__ float smu[DZ];
    __shared__ float ss2[DZ];
    __shared__ float sc[KPAD];
    __shared__ float red[256];

    const int n = blockIdx.x;
    const int t = threadIdx.x;

    if (t < DZ) { smu[t] = mu[n * DZ + t]; ss2[t] = s2[n * DZ + t]; }
    for (int k = t; k < KPAD; k += 256) sc[k] = -3.0e38f;
    __syncthreads();

    for (int k = t; k < KK; k += 256) {
        const float* Sk = S + k * DZ;
        const float* mk = m + k * DZ;
        float acc = 0.f;
#pragma unroll 8
        for (int d = 0; d < DZ; ++d) {
            float comb = ss2[d] + Sk[d];
            float diff = smu[d] - mk[d];
            acc += __logf(comb) + diff * diff / comb;
        }
        sc[k] = elp[k] - 0.5f * acc;
    }
    __syncthreads();

    red[t] = fmaxf(sc[t], sc[t + 256]);
    __syncthreads();
    for (int s = 128; s > 0; s >>= 1) {
        if (t < s) red[t] = fmaxf(red[t], red[t + s]);
        __syncthreads();
    }
    const float mx = red[0];
    __syncthreads();

    float part = 0.f;
    for (int k = t; k < KPAD; k += 256) {
        float e = (k < KK) ? __expf(sc[k] - mx) : 0.f;
        sc[k] = e;
        part += e;
    }
    red[t] = part;
    __syncthreads();
    for (int s = 128; s > 0; s >>= 1) {
        if (t < s) red[t] += red[t + s];
        __syncthreads();
    }
    const float inv = 1.f / red[0];

    for (int k = t; k < KK; k += 256) r[n * KK + k] = sc[k] * inv;
}

// ---------------------------------------------------------------- n_k = sum_n r[n,k]
__global__ void colsum_kernel(const float* __restrict__ r, float* __restrict__ n_k) {
    __shared__ float part[8][32];
    const int x = threadIdx.x;
    const int y = threadIdx.y;
    const int k = blockIdx.x * 32 + x;
    float acc = 0.f;
    if (k < KK)
        for (int n = y; n < NN; n += 8) acc += r[n * KK + k];
    part[y][x] = acc;
    __syncthreads();
    if (y == 0 && k < KK) {
        float s = 0.f;
#pragma unroll
        for (int j = 0; j < 8; ++j) s += part[j][x];
        n_k[k] = s;
    }
}

// ---------------------------------------------------------------- a_new, b_new
__global__ void ab_kernel(const float* __restrict__ n_k,
                          float* __restrict__ a_out, float* __restrict__ b_out) {
    if (threadIdx.x != 0 || blockIdx.x != 0) return;
    float suffix = 0.f;
    for (int k = KK - 1; k >= 0; --k) {
        float nk = n_k[k];
        a_out[k] = 1.f + nk;
        b_out[k] = ALPHA_C + suffix;
        suffix  += nk;
    }
}

// ---------------------------------------------------------------- pack rT[KPAD][NN] bf16 (K-major panel, zero-padded)
__global__ void __launch_bounds__(256)
pack_rT_kernel(const float* __restrict__ r, __bf16* __restrict__ rT) {
    __shared__ float tile[32][33];
    const int n0 = blockIdx.x * 32;
    const int k0 = blockIdx.y * 32;
    const int x = threadIdx.x;           // 0..31
    const int y = threadIdx.y;           // 0..7
    const int k = k0 + x;
    for (int yy = y; yy < 32; yy += 8)
        tile[yy][x] = (k < KK) ? r[(n0 + yy) * KK + k] : 0.f;   // coalesced over k
    __syncthreads();
    for (int yy = y; yy < 32; yy += 8)
        rT[(size_t)(k0 + yy) * NN + n0 + x] = (__bf16)tile[x][yy];  // coalesced over n
}

// ---------------------------------------------------------------- pack Xt[128][NN] bf16: Xt[c][n] = c<64 ? mu : s2+mu^2
__global__ void __launch_bounds__(256)
pack_Xt_kernel(const float* __restrict__ mu, const float* __restrict__ s2,
               __bf16* __restrict__ Xt) {
    __shared__ float tile[32][33];
    const int n0 = blockIdx.x * 32;
    const int c0 = blockIdx.y * 32;
    const int x = threadIdx.x;
    const int y = threadIdx.y;
    const int c = c0 + x;
    for (int yy = y; yy < 32; yy += 8) {
        const int n = n0 + yy;
        float v;
        if (c < DZ) {
            v = mu[n * DZ + c];
        } else {
            const float mm = mu[n * DZ + c - DZ];
            v = s2[n * DZ + c - DZ] + mm * mm;
        }
        tile[yy][x] = v;
    }
    __syncthreads();
    for (int yy = y; yy < 32; yy += 8)
        Xt[(size_t)(c0 + yy) * NN + n0 + x] = (__bf16)tile[x][yy];
}

// ---------------------------------------------------------------- GEMM: rt[512,128] = r^T @ [mu | s2+mu^2]
// K-major bf16 panels: each lane's WMMA fragment = two contiguous 16B global
// loads (ISA 16-bit A/B layout: K offsets {base..base+7, base+16..base+23}).
// Inner loop: 4x global_load_b128 + prefetch + v_wmma_f32_16x16x32_bf16.
__global__ void __launch_bounds__(32)
gemm_rtX_kernel(const __bf16* __restrict__ rT,   // KPAD x NN
                const __bf16* __restrict__ Xt,   // 128  x NN
                float* __restrict__ rt)          // KPAD x 128
{
    const int tile = blockIdx.x;          // 0..255
    const int tm   = tile >> 3;           // M tile
    const int tn   = tile & 7;            // N tile
    const int lane = threadIdx.x;
    const int half = lane >> 4;
    const int l15  = lane & 15;
    const int base = half * 8;            // lane-half K base per ISA layout

    const __bf16* pa = rT + (size_t)(tm * 16 + l15) * NN + base;
    const __bf16* pb = Xt + (size_t)(tn * 16 + l15) * NN + base;

    union FragU { v16bf bf; v4u u[2]; };

    v8f acc = {};
    for (int kk0 = 0; kk0 < NN; kk0 += 32) {
        FragU fa, fb;
        fa.u[0] = *(const v4u*)(pa + kk0);        // K base..base+7
        fa.u[1] = *(const v4u*)(pa + kk0 + 16);   // K base+16..base+23
        fb.u[0] = *(const v4u*)(pb + kk0);
        fb.u[1] = *(const v4u*)(pb + kk0 + 16);
        __builtin_prefetch(pa + kk0 + 64, 0, 1);
        __builtin_prefetch(pb + kk0 + 64, 0, 1);
        acc = __builtin_amdgcn_wmma_f32_16x16x32_bf16(
            /*neg_a=*/false, fa.bf, /*neg_b=*/false, fb.bf,
            /*c_mod=*/(short)0, acc, /*reuse_a=*/false, /*reuse_b=*/false);
    }

    // C/D layout: VGPR j -> M = j + half*8, N = l15
#pragma unroll
    for (int j = 0; j < 8; ++j)
        rt[(tm * 16 + j + half * 8) * 128 + tn * 16 + l15] = acc[j];
}

// ---------------------------------------------------------------- finalize m_new, s_new
__global__ void finalize_kernel(const float* __restrict__ rt,
                                const float* __restrict__ n_k,
                                float* __restrict__ m_out,
                                float* __restrict__ s_out) {
    const int i = blockIdx.x * blockDim.x + threadIdx.x;
    if (i >= KK * DZ) return;
    const int k = i >> 6;
    const int d = i & 63;
    const float nks  = fmaxf(n_k[k], 1e-10f);
    const float mnew = rt[k * 128 + d] / nks;
    const float snew = rt[k * 128 + DZ + d] / nks - mnew * mnew;
    m_out[i] = mnew;
    s_out[i] = fmaxf(snew, SMIN_C);
}

// ---------------------------------------------------------------- launch
extern "C" void kernel_launch(void* const* d_in, const int* in_sizes, int n_in,
                              void* d_out, int out_size, void* d_ws, size_t ws_size,
                              hipStream_t stream) {
    const float* mu = (const float*)d_in[0];
    const float* s2 = (const float*)d_in[1];
    const float* a  = (const float*)d_in[2];
    const float* b  = (const float*)d_in[3];
    const float* m  = (const float*)d_in[4];
    const float* S  = (const float*)d_in[5];

    float* out = (float*)d_out;
    float* r_out = out;                                  // N*K
    float* a_out = out + (size_t)NN * KK;                // K
    float* b_out = a_out + KK;                           // K
    float* m_out = b_out + KK;                           // K*DZ
    float* s_out = m_out + (size_t)KK * DZ;              // K*DZ

    float*  ws  = (float*)d_ws;
    float*  elp = ws;                                    // KPAD
    float*  n_k = ws + KPAD;                             // KPAD
    float*  rt  = ws + 2 * KPAD;                         // KPAD*128 f32
    __bf16* rT  = (__bf16*)(rt + (size_t)KPAD * 128);    // KPAD*NN bf16
    __bf16* Xt  = rT + (size_t)KPAD * NN;                // 128*NN bf16

    elogpi_kernel<<<1, 32, 0, stream>>>(a, b, elp);
    estep_kernel<<<NN, 256, 0, stream>>>(mu, s2, S, m, elp, r_out);
    colsum_kernel<<<(KK + 31) / 32, dim3(32, 8), 0, stream>>>(r_out, n_k);
    ab_kernel<<<1, 32, 0, stream>>>(n_k, a_out, b_out);
    pack_rT_kernel<<<dim3(NN / 32, KPAD / 32), dim3(32, 8), 0, stream>>>(r_out, rT);
    pack_Xt_kernel<<<dim3(NN / 32, 128 / 32), dim3(32, 8), 0, stream>>>(mu, s2, Xt);
    gemm_rtX_kernel<<<(KPAD / 16) * (128 / 16), 32, 0, stream>>>(rT, Xt, rt);
    finalize_kernel<<<(KK * DZ + 255) / 256, 256, 0, stream>>>(rt, n_k, m_out, s_out);
}
